// TEAM_77524159693403
// MI455X (gfx1250) — matrix operand order repaired and checked by reading, not compiled, not run on hardware
//
#include <hip/hip_runtime.h>

typedef __attribute__((ext_vector_type(16))) _Float16 v16h;
typedef __attribute__((ext_vector_type(8)))  _Float16 v8h;
typedef __attribute__((ext_vector_type(8)))  float    v8f;

#define HID 10
#define NSTEPS 10
#define WAVES 4                      // waves per block (128 threads, wave32)
#define ELEMS_PER_BLOCK (WAVES * 16)

struct TeamParams {
    const int*   omega;
    const float *wc, *wh, *wr;
    const float *hW1, *hb1, *hW2, *hb2, *hW3, *hb3;
    const float *Wih[2], *Whh[2], *bih[2], *bhh[2];
    const float *W1[2], *b1[2], *W2[2], *b2[2], *W3[2], *b3[2];
    int Btot;
};

// Intra-wave "sync": all scratch buffers are per-wave private and LDS ops from a
// single wave complete in order (CDNA5 ISA), so cross-lane exchange
// needs only a compile-time scheduling fence, not a hardware barrier.
__device__ __forceinline__ void wave_sync() { __builtin_amdgcn_wave_barrier(); }

// ---- fast transcendentals (v_exp_f32 + v_rcp_f32, co-execute with WMMA pipe) ----
__device__ __forceinline__ float rcp_fast(float x) { return __builtin_amdgcn_rcpf(x); }
__device__ __forceinline__ float sigm(float x)     { return rcp_fast(1.0f + __expf(-x)); }
__device__ __forceinline__ float tanh_fast(float x){ return 1.0f - 2.0f * rcp_fast(1.0f + __expf(2.0f * x)); }

// ---- WMMA fragment helpers (layouts per CDNA5 ISA §7.12.2, wave32) ----

__device__ __forceinline__ v16h frag_B(const _Float16* W, int n0, int lane) {
    const _Float16* row = W + (n0 + (lane & 15)) * 32 + ((lane >= 16) ? 8 : 0);
    v8h lo = *(const v8h*)(row);
    v8h hi = *(const v8h*)(row + 16);
    return __builtin_shufflevector(lo, hi, 0, 1, 2, 3, 4, 5, 6, 7,
                                           8, 9, 10, 11, 12, 13, 14, 15);
}

__device__ __forceinline__ v16h frag_A_ld(const _Float16* X, int lane) {
    const _Float16* b = X + (lane & 15) * 32 + ((lane >= 16) ? 8 : 0);
    v8h lo = *(const v8h*)(b);
    v8h hi = *(const v8h*)(b + 16);
    return __builtin_shufflevector(lo, hi, 0, 1, 2, 3, 4, 5, 6, 7,
                                           8, 9, 10, 11, 12, 13, 14, 15);
}

// Unguarded C stores: padding columns compute to exactly 0 (zero-padded B columns,
// zero bias, ReLU(0)=0), so storing them is harmless -> EXEC stays all-1s.
__device__ __forceinline__ void store_C48(float* dst /*[16][48]*/, v8f c, int n0, int lane,
                                          bool relu) {
    int n  = n0 + (lane & 15);
    int mb = (lane >= 16) ? 8 : 0;
#pragma unroll
    for (int g = 0; g < 8; ++g) {
        float v = c[g];
        if (relu) v = (v > 0.0f) ? v : 0.0f;
        dst[(g + mb) * 48 + n] = v;
    }
}

__device__ __forceinline__ void store_C_f16(_Float16* dst /*[16][32]*/, v8f c, int n0, int lane) {
    int n  = n0 + (lane & 15);
    int mb = (lane >= 16) ? 8 : 0;
#pragma unroll
    for (int g = 0; g < 8; ++g) {
        float v = c[g];
        v = (v > 0.0f) ? v : 0.0f;
        dst[(g + mb) * 32 + n] = (_Float16)v;
    }
}

__device__ __forceinline__ v8f wmma_f16(v16h a, v16h b, v8f c) {
    return __builtin_amdgcn_wmma_f32_16x16x32_f16(false, a, false, b, (short)0, c, false, false);
}

// ------------------------------ fused rollout ------------------------------

__global__ __launch_bounds__(128) void TEAM_77524159693403_rollout(TeamParams p,
                                                                   float* __restrict__ partial) {
    __shared__ __align__(64) _Float16 wG [2][48][32];  // gates [Wih|Whh]: N=40->48, K=12->32
    __shared__ __align__(64) _Float16 wA1[2][32][32];  // head W1: N=20->32, K=10->32
    __shared__ __align__(64) _Float16 wA2[2][32][32];  // head W2: N=20->32, K=20->32
    __shared__ float w3l[2][20];
    __shared__ __align__(16) _Float16 AS [2][WAVES][16][32]; // gates in: [s, ah, h0..h9, 0..]
    __shared__ __align__(16) _Float16 HS [2][WAVES][16][32]; // head L1 in: [h0..h9, 0..]
    __shared__ __align__(16) _Float16 A1S[WAVES][16][32];    // head L2 in: [a1(20), 0..]
    __shared__ float Cst[2][WAVES][16][HID];
    __shared__ float Gbuf[WAVES][16][48];                    // gate pre-acts / a2 (48-wide)
    __shared__ float red[128];

    const int tid = threadIdx.x;

    for (int i = tid; i < 2 * 48 * 32; i += 128) {
        int net = i / 1536, r = i % 1536, n = r / 32, k = r % 32;
        float v = 0.0f;
        if (n < 40) {
            if (k < 2)       v = p.Wih[net][n * 2 + k];
            else if (k < 12) v = p.Whh[net][n * 10 + (k - 2)];
        }
        wG[net][n][k] = (_Float16)v;
    }
    for (int i = tid; i < 2 * 32 * 32; i += 128) {
        int net = i / 1024, r = i % 1024, n = r / 32, k = r % 32;
        float v = (n < 20 && k < 10) ? p.W1[net][n * 10 + k] : 0.0f;
        wA1[net][n][k] = (_Float16)v;
    }
    for (int i = tid; i < 2 * 32 * 32; i += 128) {
        int net = i / 1024, r = i % 1024, n = r / 32, k = r % 32;
        float v = (n < 20 && k < 20) ? p.W2[net][n * 20 + k] : 0.0f;
        wA2[net][n][k] = (_Float16)v;
    }
    for (int i = tid; i < 40; i += 128) { int net = i / 20, j = i % 20; w3l[net][j] = p.W3[net][j]; }
    for (int i = tid; i < (int)(sizeof(AS) / 4);  i += 128) ((float*)AS)[i]  = 0.0f;
    for (int i = tid; i < (int)(sizeof(HS) / 4);  i += 128) ((float*)HS)[i]  = 0.0f;
    for (int i = tid; i < (int)(sizeof(A1S) / 4); i += 128) ((float*)A1S)[i] = 0.0f;
    for (int i = tid; i < 2 * WAVES * 16 * HID;   i += 128) ((float*)Cst)[i] = 0.0f;
    __syncthreads();   // cross-wave: weights staged by all threads

    const int wave = tid >> 5, lane = tid & 31;
    const int m    = lane & 15;          // element processed by this lane (dup on half-waves)
    const int jb   = lane >> 4;          // hidden-index base for pointwise (0 or 1)
    const int base = (blockIdx.x * WAVES + wave) * 16;

    float hw1[8], hb1v[4], hw2[16], hb2v[4], hw3[4], hb3v, b3r[2];
#pragma unroll
    for (int i = 0; i < 8; ++i)  hw1[i]  = p.hW1[i];
#pragma unroll
    for (int i = 0; i < 4; ++i)  hb1v[i] = p.hb1[i];
#pragma unroll
    for (int i = 0; i < 16; ++i) hw2[i]  = p.hW2[i];
#pragma unroll
    for (int i = 0; i < 4; ++i)  hb2v[i] = p.hb2[i];
#pragma unroll
    for (int i = 0; i < 4; ++i)  hw3[i]  = p.hW3[i];
    hb3v   = p.hb3[0];
    b3r[0] = p.b3[0][0];
    b3r[1] = p.b3[1][0];

    // per-element values computed by BOTH half-waves (fully uniform EXEC)
    const int  gi    = base + m;
    const bool valid = gi < p.Btot;
    const int  idx   = p.omega[valid ? gi : 0];
    const float s_star = 0.2f * (float)(idx + 1);
    const bool is04 = (idx == 1) || (idx == 3);
    const bool is24 = (idx == 0) || (idx == 4);
    const float ah_t0 = is04 ? (s_star - 0.6f) * 5.0f : 0.0f;   // /0.2
    const float ah_t4 = is24 ? (s_star - 0.6f) * 2.5f : 0.0f;   // /0.4
    float s_reg = 0.0f;

    float gbias[2][3], b1r[2][2], b2r[2][2];
#pragma unroll
    for (int net = 0; net < 2; ++net) {
#pragma unroll
        for (int j = 0; j < 3; ++j) {
            int n = j * 16 + m;
            gbias[net][j] = (n < 40) ? (p.bih[net][n] + p.bhh[net][n]) : 0.0f;
        }
#pragma unroll
        for (int j = 0; j < 2; ++j) {
            int n = j * 16 + m;
            b1r[net][j] = (n < 20) ? p.b1[net][n] : 0.0f;
            b2r[net][j] = (n < 20) ? p.b2[net][n] : 0.0f;
        }
    }

    float et = 0.0f, eh = 0.0f, er = 0.0f;

    for (int t = 0; t < NSTEPS; ++t) {
        const float tt = (float)t;

        // human MLP (all lanes, element m)
        float h1[4], h2[4];
#pragma unroll
        for (int i = 0; i < 4; ++i)
            h1[i] = tanh_fast(hw1[2 * i] * s_star + hw1[2 * i + 1] * tt + hb1v[i]);
#pragma unroll
        for (int i = 0; i < 4; ++i) {
            float acc = hb2v[i];
#pragma unroll
            for (int j = 0; j < 4; ++j) acc += hw2[4 * i + j] * h1[j];
            h2[i] = tanh_fast(acc);
        }
        float ah = hb3v;
#pragma unroll
        for (int j = 0; j < 4; ++j) ah += hw3[j] * h2[j];

        // publish [s, ah] (both half-waves write identical values -> benign, uniform EXEC)
        {
            _Float16 sh = (_Float16)s_reg, ahh = (_Float16)ah;
            AS[0][wave][m][0] = sh;  AS[0][wave][m][1] = ahh;
            AS[1][wave][m][0] = sh;  AS[1][wave][m][1] = ahh;
        }
        wave_sync();

        float onet[2];
#pragma unroll
        for (int net = 0; net < 2; ++net) {
            // LSTM gates: [16 x 12] @ [12 x 40] via 3 WMMA tiles, bias preloaded in C
            v16h A = frag_A_ld(&AS[net][wave][0][0], lane);
#pragma unroll
            for (int j = 0; j < 3; ++j) {
                v16h Bgj = frag_B(&wG[net][0][0], j * 16, lane);
                v8f  c;
#pragma unroll
                for (int g = 0; g < 8; ++g) c[g] = gbias[net][j];
                c = wmma_f16(A, Bgj, c);
                store_C48(&Gbuf[wave][0][0], c, j * 16, lane, false);
            }
            wave_sync();

            // LSTM pointwise: lane handles (m, j = jb + 2i), i = 0..4
            {
                float*    Gm = &Gbuf[wave][m][0];
                float*    Cm = &Cst[net][wave][m][0];
                _Float16* Am = &AS[net][wave][m][2];
                _Float16* Hm = &HS[net][wave][m][0];
#pragma unroll
                for (int i = 0; i < 5; ++i) {
                    int j = jb + 2 * i;
                    float gid = Gm[j];
                    float gf  = Gm[10 + j];
                    float gg  = Gm[20 + j];
                    float go  = Gm[30 + j];
                    float cc  = sigm(gf) * Cm[j] + sigm(gid) * tanh_fast(gg);
                    Cm[j] = cc;
                    _Float16 hh = (_Float16)(sigm(go) * tanh_fast(cc));
                    Am[j] = hh;
                    Hm[j] = hh;
                }
            }
            wave_sync();

            // head L1: [16 x 10] @ [10 x 20] -> ReLU -> f16 staging for L2
            v16h A2 = frag_A_ld(&HS[net][wave][0][0], lane);
#pragma unroll
            for (int j = 0; j < 2; ++j) {
                v16h B1f = frag_B(&wA1[net][0][0], j * 16, lane);
                v8f  c;
#pragma unroll
                for (int g = 0; g < 8; ++g) c[g] = b1r[net][j];
                c = wmma_f16(A2, B1f, c);
                store_C_f16(&A1S[wave][0][0], c, j * 16, lane);
            }
            wave_sync();

            // head L2: [16 x 20] @ [20 x 20] -> ReLU -> f32 (a2 -> Gbuf; no WAR on A1S)
            v16h A3 = frag_A_ld(&A1S[wave][0][0], lane);
#pragma unroll
            for (int j = 0; j < 2; ++j) {
                v16h B2f = frag_B(&wA2[net][0][0], j * 16, lane);
                v8f  c;
#pragma unroll
                for (int g = 0; g < 8; ++g) c[g] = b2r[net][j];
                c = wmma_f16(A3, B2f, c);
                store_C48(&Gbuf[wave][0][0], c, j * 16, lane, true);
            }
            wave_sync();

            // head L3: 20-wide dot (every lane computes its m)
            float o = b3r[net];
#pragma unroll
            for (int j = 0; j < 20; ++j) o += Gbuf[wave][m][j] * w3l[net][j];
            onet[net] = o;
            wave_sync();   // keep next net's/step's Gbuf stores after these reads
        }

        // advance carry + accumulate errors (all lanes; half-waves hold identical values)
        s_reg += onet[0];                       // ar from r_ net
        float ahs = (t == 0) ? ah_t0 : ((t == 4) ? ah_t4 : 0.0f);
        float de = ah - ahs;          eh += de * de;
        float dr = onet[0] - onet[1]; er += dr * dr;
        float ds = s_reg - s_star;    et += ds * ds;
    }

    // deterministic block reduction of Q (mask duplicate half-wave + OOB here only)
    float q = 0.0f;
    if (lane < 16 && valid) q = p.wc[0] * et + p.wh[0] * eh + p.wr[0] * er;
    red[tid] = q;
    __syncthreads();
#pragma unroll
    for (int s = 64; s > 0; s >>= 1) {
        if (tid < s) red[tid] += red[tid + s];
        __syncthreads();
    }
    if (tid == 0) partial[blockIdx.x] = red[0];
}

__global__ __launch_bounds__(256) void TEAM_77524159693403_reduce(const float* __restrict__ part,
                                                                  int n, float* __restrict__ out) {
    __shared__ float red[256];
    int tid = threadIdx.x;
    float v = 0.0f;
    for (int i = tid; i < n; i += 256) v += part[i];
    red[tid] = v;
    __syncthreads();
#pragma unroll
    for (int s = 128; s > 0; s >>= 1) {
        if (tid < s) red[tid] += red[tid + s];
        __syncthreads();
    }
    if (tid == 0) out[0] = red[0];
}

extern "C" void kernel_launch(void* const* d_in, const int* in_sizes, int n_in,
                              void* d_out, int out_size, void* d_ws, size_t ws_size,
                              hipStream_t stream) {
    (void)n_in; (void)out_size; (void)ws_size;
    TeamParams p{};
    p.omega = (const int*)d_in[0];
    p.wc = (const float*)d_in[1];
    p.wh = (const float*)d_in[2];
    p.wr = (const float*)d_in[3];
    p.hW1 = (const float*)d_in[4];
    p.hb1 = (const float*)d_in[5];
    p.hW2 = (const float*)d_in[6];
    p.hb2 = (const float*)d_in[7];
    p.hW3 = (const float*)d_in[8];
    p.hb3 = (const float*)d_in[9];
    int o = 10;
    for (int net = 0; net < 2; ++net) {   // net 0 = r_, net 1 = s_
        p.Wih[net] = (const float*)d_in[o + 0];
        p.Whh[net] = (const float*)d_in[o + 1];
        p.bih[net] = (const float*)d_in[o + 2];
        p.bhh[net] = (const float*)d_in[o + 3];
        p.W1[net]  = (const float*)d_in[o + 4];
        p.b1[net]  = (const float*)d_in[o + 5];
        p.W2[net]  = (const float*)d_in[o + 6];
        p.b2[net]  = (const float*)d_in[o + 7];
        p.W3[net]  = (const float*)d_in[o + 8];
        p.b3[net]  = (const float*)d_in[o + 9];
        o += 10;
    }
    const int B = in_sizes[0];
    p.Btot = B;
    const int blocks = (B + ELEMS_PER_BLOCK - 1) / ELEMS_PER_BLOCK;
    float* partial = (float*)d_ws;
    TEAM_77524159693403_rollout<<<blocks, 128, 0, stream>>>(p, partial);
    TEAM_77524159693403_reduce<<<1, 256, 0, stream>>>(partial, blocks, (float*)d_out);
}
